// InfoNCEMI_9225589752112
// MI455X (gfx1250) — compile-verified, matrix operand order checked by reading, not run on hardware
//
#include <hip/hip_runtime.h>

#define NS   8192
#define XD   512
#define HID  2048
#define YD   512

typedef __attribute__((ext_vector_type(16))) __bf16 bf16x16;
typedef __attribute__((ext_vector_type(8)))  float  f32x8;
typedef int v4i __attribute__((vector_size(16)));   // matches builtin param pointee

#if defined(__HIP_DEVICE_COMPILE__) &&                                   \
    __has_builtin(__builtin_amdgcn_global_load_async_to_lds_b128) &&     \
    __has_builtin(__builtin_amdgcn_s_wait_asynccnt)
#define USE_ASYNC 1
#else
#define USE_ASYNC 0
#endif

// 16-byte global -> LDS copy. Async path uses the CDNA5 direct-to-LDS DMA
// (tracked by ASYNCcnt); fallback is a plain register-staged copy.
// AS pointers built via integer casts: LDS aperture maps addr[31:0] to the LDS
// offset, global aperture is identity, so the truncation/extension is exact.
__device__ __forceinline__ void cp16(const void* g, void* l) {
#if USE_ASYNC
  __builtin_amdgcn_global_load_async_to_lds_b128(
      (__attribute__((address_space(1))) v4i*)(unsigned long long)(uintptr_t)g,
      (__attribute__((address_space(3))) v4i*)(unsigned int)(uintptr_t)l,
      0, 0);
#else
  *(uint4*)l = *(const uint4*)g;
#endif
}

template <int N>
__device__ __forceinline__ void wait_async() {
#if USE_ASYNC
  __builtin_amdgcn_s_wait_asynccnt(N);
#endif
}

__device__ __forceinline__ unsigned short f2bf(float x) {
  unsigned int u = __float_as_uint(x);
  u += 0x7FFFu + ((u >> 16) & 1u);          // round-to-nearest-even
  return (unsigned short)(u >> 16);
}

// Build a bf16 WMMA fragment from an LDS tile with row stride 32 bf16.
// ISA layout: lanes 0-15 take K {0..7,16..23}, lanes 16-31 take K {8..15,24..31}
//   ->  kb = (lane>=16)?8:0, contiguous 16B chunks at kb and kb+16.
__device__ __forceinline__ bf16x16 frag_from(const unsigned short* lds, int row, int kb) {
  const __bf16* p0 = (const __bf16*)(lds + row * 32 + kb);
  const __bf16* p1 = (const __bf16*)(lds + row * 32 + kb + 16);
  bf16x16 f;
#pragma unroll
  for (int i = 0; i < 8; ++i) { f[i] = p0[i]; f[i + 8] = p1[i]; }
  return f;
}

// ---------------- conversion kernels ----------------
__global__ void k_cvt_bf16(const float* __restrict__ in, unsigned short* __restrict__ out, int n) {
  int i = blockIdx.x * blockDim.x + threadIdx.x;
  if (i < n) out[i] = f2bf(in[i]);
}

// in: f32 [R x C] row-major  ->  out: bf16 [C x R] (transposed)
__global__ void k_cvt_transpose(const float* __restrict__ in, unsigned short* __restrict__ out,
                                int R, int C) {
  int o = blockIdx.x * blockDim.x + threadIdx.x;
  if (o < R * C) {
    int c = o / R, r = o - c * R;
    out[o] = f2bf(in[r * C + c]);
  }
}

// ---------------- column sums of y ----------------
__global__ void k_colsums(const float* __restrict__ y, float* __restrict__ Sy, float* __restrict__ Sy2) {
  __shared__ float s1[256], s2[256];
  int col = blockIdx.x;
  float a = 0.f, b = 0.f;
  for (int r = threadIdx.x; r < NS; r += 256) {
    float v = y[r * YD + col];
    a += v; b += v * v;
  }
  s1[threadIdx.x] = a; s2[threadIdx.x] = b;
  __syncthreads();
  for (int s = 128; s > 0; s >>= 1) {
    if ((int)threadIdx.x < s) { s1[threadIdx.x] += s1[threadIdx.x + s]; s2[threadIdx.x] += s2[threadIdx.x + s]; }
    __syncthreads();
  }
  if (threadIdx.x == 0) { Sy[col] = s1[0]; Sy2[col] = s2[0]; }
}

// ---------------- GEMM1: H = relu(X @ W1 + b1), bf16 out ----------------
// A [NS x XD] bf16 row-major, Bt [HID x XD] bf16 (W1 transposed), H [NS x HID] bf16
// Double-buffered LDS; async global->LDS copies overlap with WMMA.
__global__ __launch_bounds__(256) void k_gemm1(const unsigned short* __restrict__ A,
                                               const unsigned short* __restrict__ Bt,
                                               const float* __restrict__ bias,
                                               unsigned short* __restrict__ H) {
  __shared__ unsigned short ldsA[2][128 * 32];
  __shared__ unsigned short ldsB[2][128 * 32];
  const int tid  = threadIdx.x;
  const int wave = tid >> 5, lane = tid & 31;
  const int waveRow = wave >> 2, waveCol = wave & 3;     // 2 x 4 wave grid
  const int rowBase = blockIdx.y * 128;
  const int colBase = blockIdx.x * 128;
  const int half = lane >> 4, l16 = lane & 15;
  const int kb = half * 8;

  f32x8 acc[4][2];
#pragma unroll
  for (int r = 0; r < 4; ++r)
#pragma unroll
    for (int c = 0; c < 2; ++c)
#pragma unroll
      for (int j = 0; j < 8; ++j) acc[r][c][j] = 0.f;

  // stage one 128x32 A tile + 128x32 Bt tile into buffer `buf`
  // (512 x 16B chunks each; 2 per thread per tile -> 4 async ops / lane / buffer)
  auto stage = [&](int kk, int buf) {
#pragma unroll
    for (int c = tid; c < 512; c += 256) {
      int row = c >> 2, cq = c & 3;
      size_t gk = (size_t)(kk >> 3) + cq;
      cp16(A  + ((size_t)(rowBase + row) * XD) + gk * 8, &ldsA[buf][c * 8]);
      cp16(Bt + ((size_t)(colBase + row) * XD) + gk * 8, &ldsB[buf][c * 8]);
    }
  };

  const int NIT = XD / 32;              // 16
  stage(0, 0);
  for (int it = 0; it < NIT; ++it) {
    const int cur = it & 1;
    if (it + 1 < NIT) {
      stage((it + 1) * 32, cur ^ 1);    // prefetch next buffer (async)
      wait_async<4>();                  // current buffer's 4 ops done (in-order)
    } else {
      wait_async<0>();
    }
    __syncthreads();

    bf16x16 bfr[2];
#pragma unroll
    for (int c = 0; c < 2; ++c)
      bfr[c] = frag_from(ldsB[cur], waveCol * 32 + c * 16 + l16, kb);
#pragma unroll
    for (int r = 0; r < 4; ++r) {
      bf16x16 afr = frag_from(ldsA[cur], waveRow * 64 + r * 16 + l16, kb);
#pragma unroll
      for (int c = 0; c < 2; ++c)
        acc[r][c] = __builtin_amdgcn_wmma_f32_16x16x32_bf16(
            false, afr, false, bfr[c], (short)0, acc[r][c], false, false);
    }
    __syncthreads();                    // done reading `cur` before it is refilled
  }

  // epilogue: bias + relu -> bf16
#pragma unroll
  for (int r = 0; r < 4; ++r)
#pragma unroll
    for (int c = 0; c < 2; ++c) {
      int gcol = colBase + waveCol * 32 + c * 16 + l16;
      float bv = bias[gcol];
#pragma unroll
      for (int j = 0; j < 8; ++j) {
        int grow = rowBase + waveRow * 64 + r * 16 + half * 8 + j;
        float v = acc[r][c][j] + bv;
        v = v > 0.f ? v : 0.f;
        H[grow * HID + gcol] = f2bf(v);
      }
    }
}

// ---------------- GEMM2 fused: mu & logvar tiles + scalar epilogue ----------------
__global__ __launch_bounds__(256) void k_gemm2_fused(
    const unsigned short* __restrict__ Hmu, const unsigned short* __restrict__ Hlv,
    const unsigned short* __restrict__ Btmu, const unsigned short* __restrict__ Btlv,
    const float* __restrict__ bmu, const float* __restrict__ blv,
    const float* __restrict__ y, const float* __restrict__ Sy, const float* __restrict__ Sy2,
    float* __restrict__ partials) {
  __shared__ unsigned short lAm[2][128 * 32], lAl[2][128 * 32];
  __shared__ unsigned short lBm[2][128 * 32], lBl[2][128 * 32];
  __shared__ float red[256];

  const int tid  = threadIdx.x;
  const int wave = tid >> 5, lane = tid & 31;
  const int waveRow = wave >> 2, waveCol = wave & 3;
  const int rowBase = blockIdx.y * 128;
  const int colBase = blockIdx.x * 128;
  const int half = lane >> 4, l16 = lane & 15;
  const int kb = half * 8;

  f32x8 am[4][2], al[4][2];
#pragma unroll
  for (int r = 0; r < 4; ++r)
#pragma unroll
    for (int c = 0; c < 2; ++c)
#pragma unroll
      for (int j = 0; j < 8; ++j) { am[r][c][j] = 0.f; al[r][c][j] = 0.f; }

  // 4 tiles per buffer -> 8 async ops / lane / buffer
  auto stage = [&](int kk, int buf) {
#pragma unroll
    for (int c = tid; c < 512; c += 256) {
      int row = c >> 2, cq = c & 3;
      size_t gk = (size_t)(kk >> 3) + cq;
      cp16(Hmu  + ((size_t)(rowBase + row) * HID) + gk * 8, &lAm[buf][c * 8]);
      cp16(Hlv  + ((size_t)(rowBase + row) * HID) + gk * 8, &lAl[buf][c * 8]);
      cp16(Btmu + ((size_t)(colBase + row) * HID) + gk * 8, &lBm[buf][c * 8]);
      cp16(Btlv + ((size_t)(colBase + row) * HID) + gk * 8, &lBl[buf][c * 8]);
    }
  };

  const int NIT = HID / 32;             // 64
  stage(0, 0);
  for (int it = 0; it < NIT; ++it) {
    const int cur = it & 1;
    if (it + 1 < NIT) {
      stage((it + 1) * 32, cur ^ 1);
      wait_async<8>();
    } else {
      wait_async<0>();
    }
    __syncthreads();

    bf16x16 bm2[2], bl2[2];
#pragma unroll
    for (int c = 0; c < 2; ++c) {
      bm2[c] = frag_from(lBm[cur], waveCol * 32 + c * 16 + l16, kb);
      bl2[c] = frag_from(lBl[cur], waveCol * 32 + c * 16 + l16, kb);
    }
#pragma unroll
    for (int r = 0; r < 4; ++r) {
      bf16x16 afm = frag_from(lAm[cur], waveRow * 64 + r * 16 + l16, kb);
      bf16x16 afl = frag_from(lAl[cur], waveRow * 64 + r * 16 + l16, kb);
#pragma unroll
      for (int c = 0; c < 2; ++c) {
        am[r][c] = __builtin_amdgcn_wmma_f32_16x16x32_bf16(
            false, afm, false, bm2[c], (short)0, am[r][c], false, false);
        al[r][c] = __builtin_amdgcn_wmma_f32_16x16x32_bf16(
            false, afl, false, bl2[c], (short)0, al[r][c], false, false);
      }
    }
    __syncthreads();
  }

  // epilogue: result = -(1/N) sum (mu-y)^2 w  +  (1/N^2) sum w (N mu^2 - 2 mu Sy + Sy2)
  const float INV_N  = 1.0f / 8192.0f;
  const float INV_N2 = INV_N * INV_N;
  const float NF     = 8192.0f;
  float part = 0.f;
#pragma unroll
  for (int r = 0; r < 4; ++r)
#pragma unroll
    for (int c = 0; c < 2; ++c) {
      int gcol = colBase + waveCol * 32 + c * 16 + l16;
      float bmv = bmu[gcol], blv_ = blv[gcol];
      float syv = Sy[gcol],  sy2v = Sy2[gcol];
#pragma unroll
      for (int j = 0; j < 8; ++j) {
        int grow = rowBase + waveRow * 64 + r * 16 + half * 8 + j;
        float mu = am[r][c][j] + bmv;
        float lv = tanhf(al[r][c][j] + blv_);
        float w  = 0.5f * expf(-lv);
        float yv = y[grow * YD + gcol];
        float d  = mu - yv;
        part += -d * d * w * INV_N
              +  w * (NF * mu * mu - 2.0f * mu * syv + sy2v) * INV_N2;
      }
    }

  red[tid] = part;
  __syncthreads();
  for (int s = 128; s > 0; s >>= 1) {
    if (tid < s) red[tid] += red[tid + s];
    __syncthreads();
  }
  if (tid == 0) partials[blockIdx.y * gridDim.x + blockIdx.x] = red[0];
}

// ---------------- finalize: fixed-order sum of 256 partials ----------------
__global__ void k_finalize(const float* __restrict__ partials, float* __restrict__ out) {
  __shared__ float red[256];
  red[threadIdx.x] = partials[threadIdx.x];
  __syncthreads();
  for (int s = 128; s > 0; s >>= 1) {
    if ((int)threadIdx.x < s) red[threadIdx.x] += red[threadIdx.x + s];
    __syncthreads();
  }
  if (threadIdx.x == 0) out[0] = red[0];
}

extern "C" void kernel_launch(void* const* d_in, const int* in_sizes, int n_in,
                              void* d_out, int out_size, void* d_ws, size_t ws_size,
                              hipStream_t stream) {
  const float* x    = (const float*)d_in[0];
  const float* y    = (const float*)d_in[1];
  const float* muW1 = (const float*)d_in[2];
  const float* mub1 = (const float*)d_in[3];
  const float* muW2 = (const float*)d_in[4];
  const float* mub2 = (const float*)d_in[5];
  const float* lvW1 = (const float*)d_in[6];
  const float* lvb1 = (const float*)d_in[7];
  const float* lvW2 = (const float*)d_in[8];
  const float* lvb2 = (const float*)d_in[9];
  float* out = (float*)d_out;

  char* ws = (char*)d_ws;
  size_t off = 0;
  auto carve = [&](size_t bytes) -> char* {
    char* p = ws + off;
    off = (off + bytes + 255) & ~(size_t)255;
    return p;
  };
  unsigned short* Xb    = (unsigned short*)carve((size_t)NS * XD * 2);
  unsigned short* W1tmu = (unsigned short*)carve((size_t)HID * XD * 2);
  unsigned short* W1tlv = (unsigned short*)carve((size_t)HID * XD * 2);
  unsigned short* W2tmu = (unsigned short*)carve((size_t)YD * HID * 2);
  unsigned short* W2tlv = (unsigned short*)carve((size_t)YD * HID * 2);
  unsigned short* Hmu   = (unsigned short*)carve((size_t)NS * HID * 2);
  unsigned short* Hlv   = (unsigned short*)carve((size_t)NS * HID * 2);
  float* Sy       = (float*)carve(YD * 4);
  float* Sy2      = (float*)carve(YD * 4);
  float* partials = (float*)carve(256 * 4);
  (void)in_sizes; (void)n_in; (void)out_size; (void)ws_size;

  // 1) precision conversion + weight transposes
  k_cvt_bf16<<<(NS * XD + 255) / 256, 256, 0, stream>>>(x, Xb, NS * XD);
  k_cvt_transpose<<<(XD * HID + 255) / 256, 256, 0, stream>>>(muW1, W1tmu, XD, HID);
  k_cvt_transpose<<<(XD * HID + 255) / 256, 256, 0, stream>>>(lvW1, W1tlv, XD, HID);
  k_cvt_transpose<<<(HID * YD + 255) / 256, 256, 0, stream>>>(muW2, W2tmu, HID, YD);
  k_cvt_transpose<<<(HID * YD + 255) / 256, 256, 0, stream>>>(lvW2, W2tlv, HID, YD);

  // 2) y column statistics
  k_colsums<<<YD, 256, 0, stream>>>(y, Sy, Sy2);

  // 3) layer-1 GEMMs (bias + relu fused)
  k_gemm1<<<dim3(HID / 128, NS / 128), 256, 0, stream>>>(Xb, W1tmu, mub1, Hmu);
  k_gemm1<<<dim3(HID / 128, NS / 128), 256, 0, stream>>>(Xb, W1tlv, lvb1, Hlv);

  // 4) layer-2 GEMMs fused with the MI objective epilogue -> 256 partials
  k_gemm2_fused<<<dim3(YD / 128, NS / 128), 256, 0, stream>>>(
      Hmu, Hlv, W2tmu, W2tlv, mub2, lvb2, y, Sy, Sy2, partials);

  // 5) deterministic final reduction
  k_finalize<<<1, 256, 0, stream>>>(partials, out);
}
